// AffineModulatedLatentODEFunc_45964740002526
// MI455X (gfx1250) — matrix operand order repaired and checked by reading, not compiled, not run on hardware
//
#include <hip/hip_runtime.h>
#include <hip/hip_bf16.h>
#include <math.h>

// CDNA5 / gfx1250, wave32.
typedef __attribute__((ext_vector_type(16))) __bf16 v16bf;
typedef __attribute__((ext_vector_type(8)))  __bf16 bf16x8;
typedef __attribute__((ext_vector_type(8)))  float  v8f;

#define DE    128
#define NC    64
#define KCONV 576   // k = tap*64 + ic  (tap-major im2col order)

// ---------------------------------------------------------------------------
// Fast device math (native ops, no libm branches)
static __device__ __forceinline__ float elu1(float x) {
  return x > 0.f ? x : (__expf(x) - 1.f);
}
static __device__ __forceinline__ float tanh_fast(float x) {
#if __has_builtin(__builtin_amdgcn_tanhf)
  return __builtin_amdgcn_tanhf(x);        // gfx1250 v_tanh_f32
#else
  float e = __expf(2.f * x);
  return (e - 1.f) / (e + 1.f);
#endif
}

// ---------------------------------------------------------------------------
// Fragment helpers.  WMMA 16x16x32 bf16, wave32 layouts (CDNA5 ISA 7.12.2).
static __device__ __forceinline__ v16bf concat8(bf16x8 lo, bf16x8 hi) {
  return __builtin_shufflevector(lo, hi, 0,1,2,3,4,5,6,7,8,9,10,11,12,13,14,15);
}

// A from row-major [16][lda] LDS tile (lda multiple of 8 -> 16B aligned).
// lanes 0-15: M=lane, a[0..7]=K(k0..k0+7), a[8..15]=K(k0+16..23)
// lanes16-31: same M, K offset +8.
static __device__ __forceinline__
v16bf load_a_rm(const __bf16* A, int lda, int k0, int lane) {
  const __bf16* p = A + (lane & 15) * lda + k0 + ((lane >> 4) << 3);
  return concat8(*(const bf16x8*)p, *(const bf16x8*)(p + 16));
}

// B pre-swizzled in LDS: fragment (kt,nt) = 32 lanes x 16 contiguous bf16.
static __device__ __forceinline__
v16bf load_b_sw(const __bf16* base, int NT, int kt, int nt, int lane) {
  const __bf16* p = base + (((kt * NT + nt) << 5) + lane) * 16;
  return concat8(*(const bf16x8*)p, *(const bf16x8*)(p + 8));
}

// Stage a row-major fp32 W[K][N] into swizzled bf16 B-fragment layout.
static __device__ __forceinline__
void stage_b(const float* __restrict__ W, __bf16* dst, int KT, int NT, int N,
             int tid, int nthreads) {
  const int total = KT * NT * 512;
  for (int d = tid; d < total; d += nthreads) {
    const int kt = d / (NT * 512);
    const int r  = d % (NT * 512);
    const int nt = r >> 9, q = r & 511;
    const int lane = q >> 4, i = q & 15;
    const int k = kt * 32 + ((lane >> 4) << 3) + (i & 7) + ((i >> 3) << 4);
    const int n = nt * 16 + (lane & 15);
    dst[d] = (__bf16)W[k * N + n];
  }
}

// Same, but source is a conv weight (oc, 64, 3, 3) with k = tap*64+ic.
static __device__ __forceinline__
void stage_b_conv(const float* __restrict__ cw, __bf16* dst, int tid, int nthreads) {
  const int total = 18 * 4 * 512;          // KT=18, NT=4
  for (int d = tid; d < total; d += nthreads) {
    const int kt = d / (4 * 512);
    const int r  = d % (4 * 512);
    const int nt = r >> 9, q = r & 511;
    const int lane = q >> 4, i = q & 15;
    const int k = kt * 32 + ((lane >> 4) << 3) + (i & 7) + ((i >> 3) << 4);
    const int n = nt * 16 + (lane & 15);
    const int tap = k >> 6, ic = k & 63;
    dst[d] = (__bf16)cw[(n * 64 + ic) * 9 + tap];
  }
}

// ===========================================================================
// Kernel 1: MLP  x(32) -> elu -> h(128) -> elu -> xi(128) -> gb(256) -> d_ws
// One wave = 16 batch rows.  4 waves/block; weights swizzled in LDS.
// ===========================================================================
__global__ __launch_bounds__(128)
void mlp_kernel(const float* __restrict__ t, const float* __restrict__ mu,
                const float* __restrict__ W1, const float* __restrict__ b1,
                const float* __restrict__ W2, const float* __restrict__ b2,
                const float* __restrict__ Wg, const float* __restrict__ bg,
                float* __restrict__ gates /* B x 256 */) {
  __shared__ __attribute__((aligned(16))) __bf16 sW1[1 * 8 * 512];    //  8 KB
  __shared__ __attribute__((aligned(16))) __bf16 sW2[4 * 8 * 512];    // 32 KB
  __shared__ __attribute__((aligned(16))) __bf16 sWg[4 * 16 * 512];   // 64 KB
  __shared__ float sb1[DE], sb2[DE], sbg[256];
  __shared__ __attribute__((aligned(16))) __bf16 sx [4][16 * 32];
  __shared__ __attribute__((aligned(16))) __bf16 sh [4][16 * DE];
  __shared__ __attribute__((aligned(16))) __bf16 sxi[4][16 * DE];

  const int tid = threadIdx.x;
  stage_b(W1, sW1, 1, 8, DE, tid, 128);
  stage_b(W2, sW2, 4, 8, DE, tid, 128);
  stage_b(Wg, sWg, 4, 16, 256, tid, 128);
  if (tid < DE) { sb1[tid] = b1[tid]; sb2[tid] = b2[tid]; }
  for (int i = tid; i < 256; i += 128) sbg[i] = bg[i];
  __syncthreads();

  const int wave = tid >> 5, lane = tid & 31;
  const int base = (blockIdx.x * 4 + wave) * 16;
  const int rb   = (lane >> 4) << 3;     // C-frag row base
  const int ncol = lane & 15;            // C-frag column-in-tile

  // --- sinusoidal features: x[16][32] ---
  const float TM = 31.41592653589793f;
  float fr[4]; fr[0] = 1.f; fr[1] = 1.f / TM; fr[2] = fr[1] / TM; fr[3] = fr[2] / TM;
  __bf16* xact = sx[wave];
  for (int i = lane; i < 16 * 32; i += 32) {
    const int row = i >> 5, c = i & 31, b = base + row;
    float v; int j;
    if (c < 8) { v = t[b]; j = c; }
    else       { v = mu[b * 3 + ((c - 8) >> 3)]; j = (c - 8) & 7; }
    const float w = v * fr[j & 3];
    xact[row * 32 + c] = (__bf16)((j < 4) ? __sinf(w) : __cosf(w));
  }
  __syncthreads();

  // --- layer 1: (16x32)@(32x128), K=32 ---
  __bf16* hact = sh[wave];
#pragma unroll
  for (int nt = 0; nt < 8; ++nt) {
    v8f c = {};
    v16bf a  = load_a_rm(xact, 32, 0, lane);
    v16bf bw = load_b_sw(sW1, 8, 0, nt, lane);
    c = __builtin_amdgcn_wmma_f32_16x16x32_bf16(false, a, false, bw, (short)0, c, false, false);
    const int col = nt * 16 + ncol;
    const float bias = sb1[col];
#pragma unroll
    for (int r = 0; r < 8; ++r)
      hact[(rb + r) * DE + col] = (__bf16)elu1(c[r] + bias);
  }
  __syncthreads();

  // --- layer 2: (16x128)@(128x128), K=128 ---
  __bf16* xiact = sxi[wave];
#pragma unroll
  for (int nt = 0; nt < 8; ++nt) {
    v8f c = {};
#pragma unroll
    for (int kt = 0; kt < 4; ++kt) {
      v16bf a  = load_a_rm(hact, DE, kt * 32, lane);
      v16bf bw = load_b_sw(sW2, 8, kt, nt, lane);
      c = __builtin_amdgcn_wmma_f32_16x16x32_bf16(false, a, false, bw, (short)0, c, false, false);
    }
    const int col = nt * 16 + ncol;
    const float bias = sb2[col];
#pragma unroll
    for (int r = 0; r < 8; ++r)
      xiact[(rb + r) * DE + col] = (__bf16)elu1(c[r] + bias);
  }
  __syncthreads();

  // --- layer 3: (16x128)@(128x256) -> gates ---
#pragma unroll
  for (int nt = 0; nt < 16; ++nt) {
    v8f c = {};
#pragma unroll
    for (int kt = 0; kt < 4; ++kt) {
      v16bf a  = load_a_rm(xiact, DE, kt * 32, lane);
      v16bf bw = load_b_sw(sWg, 16, kt, nt, lane);
      c = __builtin_amdgcn_wmma_f32_16x16x32_bf16(false, a, false, bw, (short)0, c, false, false);
    }
    const int col = nt * 16 + ncol;
    const float bias = sbg[col];
#pragma unroll
    for (int r = 0; r < 8; ++r)
      gates[(size_t)(base + rb + r) * 256 + col] = c[r] + bias;
  }
}

// ===========================================================================
// Kernel 2: conv stack.  One wave = one item.  conv1/conv2 as im2col GEMM:
//   M=16 px, K=576 (tap*64+ic, zero row for borders), N=64 oc.
// Activations transposed in LDS: hh[px][ic] (+ zero row at px=16) so every
// A half-fragment is one 16B contiguous ds_load_b128.
// ===========================================================================
static __device__ __forceinline__
bf16x8 conv_a_half(const __bf16* hhT, int kk, int px, const signed char* smap) {
  const int tap = kk >> 6, icb = kk & 63;
  const int s = smap[px * 9 + tap];        // 16 == zero row
  return *(const bf16x8*)(hhT + s * 64 + icb);
}

static __device__ __forceinline__
v16bf load_a_conv(const __bf16* hhT, int k0, int lane, const signed char* smap) {
  const int px = lane & 15;
  const int kb = k0 + ((lane >> 4) << 3);
  return concat8(conv_a_half(hhT, kb, px, smap),
                 conv_a_half(hhT, kb + 16, px, smap));
}

static __device__ __forceinline__
void conv_layer(const __bf16* hinT, const __bf16* Bw,
                const float* __restrict__ cbias, const float* __restrict__ g,
                int which, __bf16* houtT, int lane, const signed char* smap) {
  const int rbase = (lane >> 4) << 3;
#pragma unroll
  for (int nt = 0; nt < 4; ++nt) {
    v8f c = {};
#pragma unroll
    for (int kt = 0; kt < 18; ++kt) {
      v16bf a  = load_a_conv(hinT, kt * 32, lane, smap);
      v16bf bw = load_b_sw(Bw, 4, kt, nt, lane);
      c = __builtin_amdgcn_wmma_f32_16x16x32_bf16(false, a, false, bw, (short)0, c, false, false);
    }
    const int oc = nt * 16 + (lane & 15);
    const float gg = g[which * NC + oc];
    const float bb = g[(which + 2) * NC + oc];
    const float cb = cbias[oc];
#pragma unroll
    for (int r = 0; r < 8; ++r)
      houtT[(rbase + r) * 64 + oc] = (__bf16)tanh_fast(gg * (c[r] + cb) + bb);
  }
}

__global__ __launch_bounds__(64)
void conv_kernel(const float* __restrict__ z,
                 const float* __restrict__ cin_w, const float* __restrict__ cin_b,
                 const float* __restrict__ c1_w,  const float* __restrict__ c1_b,
                 const float* __restrict__ c2_w,  const float* __restrict__ c2_b,
                 const float* __restrict__ cout_w, const float* __restrict__ cout_b,
                 const float* __restrict__ gates, float* __restrict__ out,
                 int nwaves_total, int iters) {
  __shared__ __attribute__((aligned(16))) __bf16 sB1[18 * 4 * 512];  // 72 KB
  __shared__ __attribute__((aligned(16))) __bf16 sB2[18 * 4 * 512];  // 72 KB
  __shared__ __bf16 scin [NC * 9];             // cin_w  [oc][tap]
  __shared__ __bf16 scout[NC * 9];             // cout_w [ic][tap]
  __shared__ signed char ssmap[16 * 9];        // (px,tap) -> src px, 16 = zero
  __shared__ float  szbuf[2][17];              // [16] = 0
  __shared__ __attribute__((aligned(16))) __bf16 shhA[2][17 * 64];   // row 16 = 0
  __shared__ __attribute__((aligned(16))) __bf16 shhB[2][17 * 64];

  const int tid = threadIdx.x;
  stage_b_conv(c1_w, sB1, tid, 64);
  stage_b_conv(c2_w, sB2, tid, 64);
  for (int i = tid; i < NC * 9; i += 64) {
    scin [i] = (__bf16)cin_w[i];
    scout[i] = (__bf16)cout_w[i];
  }
  for (int i = tid; i < 16 * 9; i += 64) {
    const int px = i / 9, tap = i % 9;
    const int sy = (px >> 2) + tap / 3 - 1, sx = (px & 3) + tap % 3 - 1;
    ssmap[i] = (sy >= 0 && sy < 4 && sx >= 0 && sx < 4)
                 ? (signed char)(sy * 4 + sx) : (signed char)16;
  }
  // zero rows (never rewritten)
  for (int i = tid; i < 64; i += 64) {
    shhA[0][16 * 64 + i] = (__bf16)0.0f; shhA[1][16 * 64 + i] = (__bf16)0.0f;
    shhB[0][16 * 64 + i] = (__bf16)0.0f; shhB[1][16 * 64 + i] = (__bf16)0.0f;
  }
  if (tid < 2) szbuf[tid][16] = 0.f;
  __syncthreads();

  const int wave = tid >> 5, lane = tid & 31;
  const int gwave = blockIdx.x * 2 + wave;

  for (int it = 0; it < iters; ++it) {         // uniform across all waves
    const int item = gwave + it * nwaves_total;
    const float* g = gates + (size_t)item * 256;

    if (lane < 16) szbuf[wave][lane] = z[(size_t)item * 16 + lane];
    __syncthreads();

    // conv_in (1->64) + tanh, scalar (<1% of FLOPs) -> hh[px][oc]
    for (int i = lane; i < NC * 16; i += 32) {
      const int oc = i >> 4, px = i & 15;
      float acc = cin_b[oc];
#pragma unroll
      for (int tap = 0; tap < 9; ++tap)
        acc += (float)scin[oc * 9 + tap] * szbuf[wave][(int)ssmap[px * 9 + tap]];
      shhA[wave][px * 64 + oc] = (__bf16)tanh_fast(acc);
    }
    __syncthreads();

    conv_layer(shhA[wave], sB1, c1_b, g, 0, shhB[wave], lane, ssmap);
    __syncthreads();
    conv_layer(shhB[wave], sB2, c2_b, g, 1, shhA[wave], lane, ssmap);
    __syncthreads();

    // conv_out (64->1), scalar
    if (lane < 16) {
      float acc = cout_b[0];
      for (int ic = 0; ic < NC; ++ic)
#pragma unroll
        for (int tap = 0; tap < 9; ++tap)
          acc += (float)scout[ic * 9 + tap]
               * (float)shhA[wave][(int)ssmap[lane * 9 + tap] * 64 + ic];
      out[(size_t)item * 16 + lane] = acc;
    }
    __syncthreads();
  }
}

// ===========================================================================
extern "C" void kernel_launch(void* const* d_in, const int* in_sizes, int n_in,
                              void* d_out, int out_size, void* d_ws, size_t ws_size,
                              hipStream_t stream) {
  const float* t      = (const float*)d_in[0];
  const float* z      = (const float*)d_in[1];
  const float* mu     = (const float*)d_in[2];
  const float* W1     = (const float*)d_in[3];
  const float* b1     = (const float*)d_in[4];
  const float* W2     = (const float*)d_in[5];
  const float* b2     = (const float*)d_in[6];
  const float* Wg     = (const float*)d_in[7];
  const float* bg     = (const float*)d_in[8];
  const float* cin_w  = (const float*)d_in[9];
  const float* cin_b  = (const float*)d_in[10];
  const float* c1_w   = (const float*)d_in[11];
  const float* c1_b   = (const float*)d_in[12];
  const float* c2_w   = (const float*)d_in[13];
  const float* c2_b   = (const float*)d_in[14];
  const float* cout_w = (const float*)d_in[15];
  const float* cout_b = (const float*)d_in[16];
  float* outp  = (float*)d_out;
  float* gates = (float*)d_ws;            // needs B*256*4 = 32 MB scratch

  const int B = in_sizes[0];              // 32768

  mlp_kernel<<<B / 64, 128, 0, stream>>>(t, mu, W1, b1, W2, b2, Wg, bg, gates);

  const int nWG = 1024;
  const int nwaves = nWG * 2;             // 2048 waves
  const int iters = B / nwaves;           // 16 items/wave, uniform
  conv_kernel<<<nWG, 64, 0, stream>>>(z, cin_w, cin_b, c1_w, c1_b,
                                      c2_w, c2_b, cout_w, cout_b,
                                      gates, outp, nwaves, iters);
}
